// BiasedAxialAttention_56873956933734
// MI455X (gfx1250) — matrix operand order repaired and checked by reading, not compiled, not run on hardware
//
#include <hip/hip_runtime.h>
#include <hip/hip_bf16.h>
#include <math.h>

// ---------------------------------------------------------------------------
// BiasedAxialAttention for MI455X (gfx1250), wave32 + v_wmma_f32_16x16x32_f16.
// All GEMM intermediates are f16; layouts are chosen so every WMMA fragment
// is two contiguous 16B loads per lane (see CDNA5 ISA 7.12.2).
// Hot K-loops are software-pipelined (double-buffered fragments) and issue
// global_prefetch for the next 512B of each stream.
// ---------------------------------------------------------------------------

typedef __attribute__((ext_vector_type(16))) _Float16 v16h;
typedef __attribute__((ext_vector_type(8)))  _Float16 v8h;
typedef __attribute__((ext_vector_type(8)))  float    v8f;

#define LL      256
#define DPAIR   128
#define NHEAD   8
#define DHEAD   32
#define HD      256      // NHEAD*DHEAD
#define MROWS   65536    // LL*LL
#define KC      8192     // LL*DHEAD : tied-attention contraction length
#define PFDIST  256      // prefetch distance in halves (8 K-steps = 512B)

// ---- WMMA fragment loaders -------------------------------------------------
// A (16x32, f16): lane l holds row m0 + (l&15); VGPR pairs hold
//   K = {2v,2v+1} (+8 for lanes>=16) for v<4, K = {16+..} for v>=4.
// => per lane: b128 @ (c0 + hi*8) -> halves 0..7, b128 @ (+16) -> halves 8..15
__device__ __forceinline__ v16h load_a16x32(const _Float16* __restrict__ base,
                                            int row0, int c0, int ld, int lane) {
  int r  = lane & 15;
  int hi = (lane >> 4) & 1;
  const _Float16* p = base + (size_t)(row0 + r) * ld + c0 + hi * 8;
  v8h lo = *(const v8h*)(p);
  v8h hh = *(const v8h*)(p + 16);
  v16h o;
#pragma unroll
  for (int t = 0; t < 8; ++t) { o[t] = lo[t]; o[8 + t] = hh[t]; }
  return o;
}

// B (32x16, f16) from a [col][K] layout: lane l holds column col0 + (l&15);
// same K order as A: halves 0..7 = K c0+hi*8.., halves 8..15 = K c0+16+hi*8..
__device__ __forceinline__ v16h load_b32x16(const _Float16* __restrict__ base,
                                            int col0, int c0, int ld, int lane) {
  int n  = lane & 15;
  int hi = (lane >> 4) & 1;
  const _Float16* p = base + (size_t)(col0 + n) * ld + c0 + hi * 8;
  v8h a = *(const v8h*)(p);
  v8h b = *(const v8h*)(p + 16);
  v16h o;
#pragma unroll
  for (int t = 0; t < 8; ++t) { o[t] = a[t]; o[8 + t] = b[t]; }
  return o;
}

__device__ __forceinline__ v8f wmma16(v16h a, v16h b, v8f c) {
  return __builtin_amdgcn_wmma_f32_16x16x32_f16(false, a, false, b,
                                                (short)0, c, false, false);
}

// ---- 1) weight transposes (f32 -> f16, [in][out] -> [out][in]) -------------
__global__ void k_prep_weights(const float* __restrict__ Wq, const float* __restrict__ Wk,
                               const float* __restrict__ Wv, const float* __restrict__ Wg,
                               const float* __restrict__ Wb, const float* __restrict__ Wo,
                               _Float16* WqT, _Float16* WkT, _Float16* WvT,
                               _Float16* WgT, _Float16* WbT, _Float16* WoT) {
  int tid = blockIdx.x * blockDim.x + threadIdx.x;
  if (tid < 32768) {                 // [256][128] <- [128][256]
    int o = tid >> 7, i = tid & 127;
    WqT[tid] = (_Float16)Wq[i * 256 + o];
    WkT[tid] = (_Float16)Wk[i * 256 + o];
    WvT[tid] = (_Float16)Wv[i * 256 + o];
    WgT[tid] = (_Float16)Wg[i * 256 + o];
    int dp = tid >> 8, hd = tid & 255;   // WoT[128][256] <- Wo[256][128]
    WoT[tid] = (_Float16)Wo[hd * 128 + dp];
  }
  if (tid < 1024) {                  // WbT[8][128] <- Wb[128][8]
    int h = tid >> 7, i = tid & 127;
    WbT[tid] = (_Float16)Wb[i * 8 + h];
  }
}

// ---- 2) transpose + LayerNorm, f32 -> f16 (one wave per row of 128) -------
__global__ void k_ln(const float* __restrict__ pair, const float* __restrict__ bias,
                     const float* __restrict__ wp, const float* __restrict__ bp,
                     const float* __restrict__ wb, const float* __restrict__ bb,
                     _Float16* __restrict__ PLN, _Float16* __restrict__ BLN) {
  int rid  = blockIdx.x * 8 + threadIdx.y;   // 0..131071
  int lane = threadIdx.x;
  bool isPair = rid < MROWS;
  int m = isPair ? rid : rid - MROWS;        // m = x*256+y in transposed frame
  int x = m >> 8, y = m & 255;
  const float* src = (isPair ? pair : bias) + (size_t)((y << 8) + x) * DPAIR;
  float v[4], s = 0.f, ss = 0.f;
#pragma unroll
  for (int t = 0; t < 4; ++t) {
    v[t] = src[lane + 32 * t];
    s += v[t]; ss += v[t] * v[t];
  }
#pragma unroll
  for (int off = 16; off >= 1; off >>= 1) {
    s  += __shfl_xor(s,  off, 32);
    ss += __shfl_xor(ss, off, 32);
  }
  float mean = s * (1.f / 128.f);
  float inv  = rsqrtf(ss * (1.f / 128.f) - mean * mean + 1e-5f);
  const float* w  = isPair ? wp : wb;
  const float* bv = isPair ? bp : bb;
  _Float16* dst = (isPair ? PLN : BLN) + (size_t)m * DPAIR;
#pragma unroll
  for (int t = 0; t < 4; ++t) {
    int d = lane + 32 * t;
    dst[d] = (_Float16)((v[t] - mean) * inv * w[d] + bv[d]);
  }
}

// ---- 3) fused Q/K/V/gate projection GEMM; scatter to attention layouts ----
// M=65536, K=128, N=4*256.  One 16x16 tile per wave, 4 WMMAs over K.
__global__ void k_proj(const _Float16* __restrict__ PLN,
                       const _Float16* __restrict__ WqT, const _Float16* __restrict__ WkT,
                       const _Float16* __restrict__ WvT, const _Float16* __restrict__ WgT,
                       const float* __restrict__ bg,
                       _Float16* __restrict__ Qs, _Float16* __restrict__ Ks,
                       _Float16* __restrict__ Vs, _Float16* __restrict__ G16) {
  int tile = blockIdx.x * 8 + threadIdx.y;    // 0..262143
  int lane = threadIdx.x;
  int mt = tile >> 6, nt = tile & 63;
  int m0 = mt << 4;
  int seg = nt >> 4;                          // 0=Q 1=K 2=V 3=gate
  int n0  = (nt & 15) << 4;
  const _Float16* WT = seg == 0 ? WqT : seg == 1 ? WkT : seg == 2 ? WvT : WgT;
  v8f acc = {};
#pragma unroll
  for (int c0 = 0; c0 < DPAIR; c0 += 32) {
    v16h a = load_a16x32(PLN, m0, c0, DPAIR, lane);
    v16h b = load_b32x16(WT,  n0, c0, DPAIR, lane);
    acc = wmma16(a, b, acc);
  }
  int hi  = (lane >> 4) & 1;
  int col = n0 + (lane & 15);
  int h = col >> 5, d = col & 31;
#pragma unroll
  for (int r = 0; r < 8; ++r) {
    int m = m0 + r + hi * 8;                  // m = x*256+y (x=n, y=i/j/k)
    int x = m >> 8, y = m & 255;
    float val = acc[r];
    if (seg == 0) {        // q * 1/sqrt(DH) -> Qs[h][i][n*32+d]
      Qs[(size_t)((h << 8) + y) * KC + (x << 5) + d] =
          (_Float16)(val * 0.17677669529663687f);
    } else if (seg == 1) { // k * 1/sqrt(L)  -> Ks[h][j][n*32+d]
      Ks[(size_t)((h << 8) + y) * KC + (x << 5) + d] = (_Float16)(val * 0.0625f);
    } else if (seg == 2) { // v -> Vs[h][k*32+d][j]
      Vs[(size_t)((h << 13) + (x << 5) + d) * 256 + y] = (_Float16)val;
    } else {               // gate = sigmoid(. + bg) -> G16[m][col]
      float g = 1.f / (1.f + __expf(-(val + bg[col])));
      G16[(size_t)m * HD + col] = (_Float16)g;
    }
  }
}

// ---- 4) bias projection: Bb[m][h] = BLN[m,:] . WbT[h,:]  (N=8, VALU dot) ---
__global__ void k_biasproj(const _Float16* __restrict__ BLN,
                           const _Float16* __restrict__ WbT,
                           float* __restrict__ Bb) {
  int tid = blockIdx.x * blockDim.x + threadIdx.x;   // 0..524287
  int m = tid >> 3, h = tid & 7;
  const _Float16* a = BLN + (size_t)m * DPAIR;
  const _Float16* w = WbT + h * DPAIR;
  float s = 0.f;
#pragma unroll
  for (int i = 0; i < DPAIR; i += 8) {
    v8h av = *(const v8h*)(a + i);
    v8h wv = *(const v8h*)(w + i);
#pragma unroll
    for (int t = 0; t < 8; ++t) s += (float)av[t] * (float)wv[t];
  }
  Bb[tid] = s;
}

// ---- 5) tied attention: attn[h][i][j] = sum_{n,d} q*k + b - 1000*!mask ----
// Per head: M=N=256, K=8192.  Wave computes a 16x64 strip (A reused 4x).
// Software-pipelined: fragments for step c0+32 are in flight while the WMMAs
// for step c0 execute; global_prefetch runs 8 steps (512B) ahead per stream.
__global__ void k_attn(const _Float16* __restrict__ Qs, const _Float16* __restrict__ Ks,
                       const float* __restrict__ Bb, const unsigned char* __restrict__ mask,
                       float* __restrict__ AttnF) {
  int tile = blockIdx.x * 8 + threadIdx.y;    // 0..511
  int lane = threadIdx.x;
  int h   = tile >> 6;
  int rem = tile & 63;
  int i0  = (rem >> 2) << 4;
  int j0  = (rem & 3) << 6;
  const _Float16* Qh = Qs + (size_t)h * 256 * KC;
  const _Float16* Kh = Ks + (size_t)h * 256 * KC;
  // per-lane stream base rows for prefetching
  const _Float16* pfA = Qh + (size_t)(i0 + (lane & 15)) * KC;
  const _Float16* pfB = Kh + (size_t)(j0 + ((lane & 15) << 2)) * KC; // 4 of 64 cols/lane
  v8f acc0 = {}, acc1 = {}, acc2 = {}, acc3 = {};
  v16h a  = load_a16x32(Qh, i0, 0, KC, lane);
  v16h b0 = load_b32x16(Kh, j0,      0, KC, lane);
  v16h b1 = load_b32x16(Kh, j0 + 16, 0, KC, lane);
  v16h b2 = load_b32x16(Kh, j0 + 32, 0, KC, lane);
  v16h b3 = load_b32x16(Kh, j0 + 48, 0, KC, lane);
  for (int c0 = 32; c0 < KC; c0 += 32) {
    if ((c0 & (PFDIST - 1)) == 0 && c0 + PFDIST <= KC) {
      __builtin_prefetch(pfA + c0 + PFDIST, 0, 1);   // global_prefetch_b8
      __builtin_prefetch(pfB + c0 + PFDIST, 0, 1);
    }
    v16h an  = load_a16x32(Qh, i0, c0, KC, lane);
    v16h bn0 = load_b32x16(Kh, j0,      c0, KC, lane);
    v16h bn1 = load_b32x16(Kh, j0 + 16, c0, KC, lane);
    v16h bn2 = load_b32x16(Kh, j0 + 32, c0, KC, lane);
    v16h bn3 = load_b32x16(Kh, j0 + 48, c0, KC, lane);
    acc0 = wmma16(a, b0, acc0);
    acc1 = wmma16(a, b1, acc1);
    acc2 = wmma16(a, b2, acc2);
    acc3 = wmma16(a, b3, acc3);
    a = an; b0 = bn0; b1 = bn1; b2 = bn2; b3 = bn3;
  }
  acc0 = wmma16(a, b0, acc0);
  acc1 = wmma16(a, b1, acc1);
  acc2 = wmma16(a, b2, acc2);
  acc3 = wmma16(a, b3, acc3);
  int hi = (lane >> 4) & 1;
#pragma unroll
  for (int u = 0; u < 4; ++u) {
    v8f acc = u == 0 ? acc0 : u == 1 ? acc1 : u == 2 ? acc2 : acc3;
    int j = j0 + u * 16 + (lane & 15);
#pragma unroll
    for (int r = 0; r < 8; ++r) {
      int i = i0 + r + hi * 8;
      size_t ij = (size_t)i * 256 + j;
      float v = acc[r] + Bb[ij * 8 + h];
      v -= mask[ij] ? 0.f : 1000.f;
      AttnF[(size_t)h * MROWS + ij] = v;
    }
  }
}

// ---- 6) softmax over j, f32 -> f16  (one wave per (h,i) row of 256) -------
__global__ void k_softmax(const float* __restrict__ AttnF, _Float16* __restrict__ A16) {
  int row  = blockIdx.x * 8 + threadIdx.y;    // 0..2047
  int lane = threadIdx.x;
  const float* src = AttnF + (size_t)row * 256;
  float v[8], mx = -3.4e38f;
#pragma unroll
  for (int t = 0; t < 8; ++t) { v[t] = src[lane + 32 * t]; mx = fmaxf(mx, v[t]); }
#pragma unroll
  for (int off = 16; off >= 1; off >>= 1) mx = fmaxf(mx, __shfl_xor(mx, off, 32));
  float s = 0.f;
#pragma unroll
  for (int t = 0; t < 8; ++t) { v[t] = __expf(v[t] - mx); s += v[t]; }
#pragma unroll
  for (int off = 16; off >= 1; off >>= 1) s += __shfl_xor(s, off, 32);
  float inv = 1.f / s;
  _Float16* dst = A16 + (size_t)row * 256;
#pragma unroll
  for (int t = 0; t < 8; ++t) dst[lane + 32 * t] = (_Float16)(v[t] * inv);
}

// ---- 7) out[i][k*32+d] = sum_j attn[i][j] * Vs[h][k*32+d][j], fused gate --
// Per head: M=256, N=8192, K=256.  Wave computes a 16x64 strip, pipelined.
__global__ void k_out(const _Float16* __restrict__ A16, const _Float16* __restrict__ Vs,
                      const _Float16* __restrict__ G16, _Float16* __restrict__ OutPre) {
  int tile = blockIdx.x * 8 + threadIdx.y;    // 0..16383
  int lane = threadIdx.x;
  int h   = tile >> 11;
  int rem = tile & 2047;
  int i0   = (rem >> 7) << 4;
  int col0 = (rem & 127) << 6;
  const _Float16* Ah = A16 + (size_t)h * MROWS;
  const _Float16* Vh = Vs  + (size_t)h * KC * 256;
  v8f acc0 = {}, acc1 = {}, acc2 = {}, acc3 = {};
  v16h a  = load_a16x32(Ah, i0, 0, 256, lane);
  v16h b0 = load_b32x16(Vh, col0,      0, 256, lane);
  v16h b1 = load_b32x16(Vh, col0 + 16, 0, 256, lane);
  v16h b2 = load_b32x16(Vh, col0 + 32, 0, 256, lane);
  v16h b3 = load_b32x16(Vh, col0 + 48, 0, 256, lane);
#pragma unroll
  for (int c0 = 32; c0 < 256; c0 += 32) {
    v16h an  = load_a16x32(Ah, i0, c0, 256, lane);
    v16h bn0 = load_b32x16(Vh, col0,      c0, 256, lane);
    v16h bn1 = load_b32x16(Vh, col0 + 16, c0, 256, lane);
    v16h bn2 = load_b32x16(Vh, col0 + 32, c0, 256, lane);
    v16h bn3 = load_b32x16(Vh, col0 + 48, c0, 256, lane);
    acc0 = wmma16(a, b0, acc0);
    acc1 = wmma16(a, b1, acc1);
    acc2 = wmma16(a, b2, acc2);
    acc3 = wmma16(a, b3, acc3);
    a = an; b0 = bn0; b1 = bn1; b2 = bn2; b3 = bn3;
  }
  acc0 = wmma16(a, b0, acc0);
  acc1 = wmma16(a, b1, acc1);
  acc2 = wmma16(a, b2, acc2);
  acc3 = wmma16(a, b3, acc3);
  int hi = (lane >> 4) & 1;
#pragma unroll
  for (int u = 0; u < 4; ++u) {
    v8f acc = u == 0 ? acc0 : u == 1 ? acc1 : u == 2 ? acc2 : acc3;
    int colg = col0 + u * 16 + (lane & 15);
    int k = colg >> 5, d = colg & 31;
#pragma unroll
    for (int r = 0; r < 8; ++r) {
      int i = i0 + r + hi * 8;
      size_t midx = (size_t)((i << 8) + k) * HD + (h << 5) + d;
      OutPre[midx] = (_Float16)(acc[r] * (float)G16[midx]);
    }
  }
}

// ---- 8) final projection + bo + transpose back ----------------------------
// M=65536, K=256, N=128.  d_out[(k*256+i)*128+dp] (undo the axial transpose).
__global__ void k_final(const _Float16* __restrict__ OutPre,
                        const _Float16* __restrict__ WoT,
                        const float* __restrict__ bo, float* __restrict__ out) {
  int tile = blockIdx.x * 8 + threadIdx.y;    // 0..32767
  int lane = threadIdx.x;
  int m0 = (tile >> 3) << 4;
  int n0 = (tile & 7) << 4;
  v8f acc = {};
#pragma unroll
  for (int c0 = 0; c0 < 256; c0 += 32) {
    v16h a = load_a16x32(OutPre, m0, c0, 256, lane);
    v16h b = load_b32x16(WoT,    n0, c0, 256, lane);
    acc = wmma16(a, b, acc);
  }
  int hi = (lane >> 4) & 1;
  int dp = n0 + (lane & 15);
  float bov = bo[dp];
#pragma unroll
  for (int r = 0; r < 8; ++r) {
    int m = m0 + r + hi * 8;
    int i = m >> 8, k = m & 255;
    out[((size_t)k * 256 + i) * DPAIR + dp] = acc[r] + bov;
  }
}

// ---------------------------------------------------------------------------
extern "C" void kernel_launch(void* const* d_in, const int* in_sizes, int n_in,
                              void* d_out, int out_size, void* d_ws, size_t ws_size,
                              hipStream_t stream) {
  const float* pair = (const float*)d_in[0];
  const float* bias = (const float*)d_in[1];
  const unsigned char* mask = (const unsigned char*)d_in[2];
  const float* ln_pair_w = (const float*)d_in[3];
  const float* ln_pair_b = (const float*)d_in[4];
  const float* ln_bias_w = (const float*)d_in[5];
  const float* ln_bias_b = (const float*)d_in[6];
  const float* Wq = (const float*)d_in[7];
  const float* Wk = (const float*)d_in[8];
  const float* Wv = (const float*)d_in[9];
  const float* Wb = (const float*)d_in[10];
  const float* Wg = (const float*)d_in[11];
  const float* bg = (const float*)d_in[12];
  const float* Wo = (const float*)d_in[13];
  const float* bo = (const float*)d_in[14];
  float* out = (float*)d_out;

  // workspace carve-up (~207 MB, all 256B-aligned chunks)
  char* ws = (char*)d_ws;
  size_t off = 0;
  auto alloc = [&](size_t bytes) { char* p = ws + off; off += (bytes + 255) & ~(size_t)255; return p; };
  _Float16* PLN    = (_Float16*)alloc((size_t)MROWS * DPAIR * 2);   // 16.8 MB
  _Float16* BLN    = (_Float16*)alloc((size_t)MROWS * DPAIR * 2);   // 16.8 MB
  _Float16* Qs     = (_Float16*)alloc((size_t)NHEAD * 256 * KC * 2); // 33.6 MB
  _Float16* Ks     = (_Float16*)alloc((size_t)NHEAD * 256 * KC * 2); // 33.6 MB
  _Float16* Vs     = (_Float16*)alloc((size_t)NHEAD * KC * 256 * 2); // 33.6 MB
  _Float16* G16    = (_Float16*)alloc((size_t)MROWS * HD * 2);       // 33.6 MB
  _Float16* OutPre = (_Float16*)alloc((size_t)MROWS * HD * 2);       // 33.6 MB
  float*    AttnF  = (float*)   alloc((size_t)NHEAD * MROWS * 4);    // 2.1 MB
  _Float16* A16    = (_Float16*)alloc((size_t)NHEAD * MROWS * 2);    // 1.0 MB
  float*    Bb     = (float*)   alloc((size_t)MROWS * NHEAD * 4);    // 2.1 MB
  _Float16* WqT    = (_Float16*)alloc(32768 * 2);
  _Float16* WkT    = (_Float16*)alloc(32768 * 2);
  _Float16* WvT    = (_Float16*)alloc(32768 * 2);
  _Float16* WgT    = (_Float16*)alloc(32768 * 2);
  _Float16* WoT    = (_Float16*)alloc(32768 * 2);
  _Float16* WbT    = (_Float16*)alloc(1024 * 2);

  dim3 wblk(32, 8);   // 8 wave32 waves per workgroup

  k_prep_weights<<<128, 256, 0, stream>>>(Wq, Wk, Wv, Wg, Wb, Wo,
                                          WqT, WkT, WvT, WgT, WbT, WoT);
  k_ln<<<16384, wblk, 0, stream>>>(pair, bias, ln_pair_w, ln_pair_b,
                                   ln_bias_w, ln_bias_b, PLN, BLN);
  k_proj<<<32768, wblk, 0, stream>>>(PLN, WqT, WkT, WvT, WgT, bg, Qs, Ks, Vs, G16);
  k_biasproj<<<2048, 256, 0, stream>>>(BLN, WbT, Bb);
  k_attn<<<64, wblk, 0, stream>>>(Qs, Ks, Bb, mask, AttnF);
  k_softmax<<<256, wblk, 0, stream>>>(AttnF, A16);
  k_out<<<2048, wblk, 0, stream>>>(A16, Vs, G16, OutPre);
  k_final<<<4096, wblk, 0, stream>>>(OutPre, WoT, bo, out);
}
